// SparseMambaNet_32341103739482
// MI455X (gfx1250) — matrix operand order repaired and checked by reading, not compiled, not run on hardware
//
#include <hip/hip_runtime.h>
#include <hip/hip_bf16.h>
#include <math.h>

// ---------------- types ----------------
typedef __attribute__((ext_vector_type(16))) __bf16 v16bf;
typedef __attribute__((ext_vector_type(8)))  __bf16 v8bf;
typedef __attribute__((ext_vector_type(8)))  float  v8f;
typedef __attribute__((ext_vector_type(4)))  unsigned short us4;

__device__ __forceinline__ unsigned short f2bf_u(float f) {
    unsigned u = __builtin_bit_cast(unsigned, f);
    unsigned r = u + 0x7FFFu + ((u >> 16) & 1u);     // round-to-nearest-even
    return (unsigned short)(r >> 16);
}
__device__ __forceinline__ float silu_f(float x) { return x / (1.f + __expf(-x)); }

// fragment loaders (bf16 in memory, ISA 7.12.2 layouts)
// A 16x32: lanes 0-15 K 0..7 & 16..23 ; lanes 16-31 K 8..15 & 24..31  (p already at k0+koffA)
__device__ __forceinline__ v16bf ld_frag_a(const unsigned short* p) {
    v8bf lo = *(const v8bf*)(const void*)(p);
    v8bf hi = *(const v8bf*)(const void*)(p + 16);
    return __builtin_shufflevector(lo, hi, 0,1,2,3,4,5,6,7,8,9,10,11,12,13,14,15);
}
// B 32x16: lanes 0-15 K 0..7 & 16..23 ; lanes 16-31 K 8..15 & 24..31  (p already at k0+koffB)
__device__ __forceinline__ v16bf ld_frag_b(const unsigned short* p) {
    v8bf lo = *(const v8bf*)(const void*)(p);
    v8bf hi = *(const v8bf*)(const void*)(p + 16);
    return __builtin_shufflevector(lo, hi, 0,1,2,3,4,5,6,7,8,9,10,11,12,13,14,15);
}

// ---------------- f32 -> bf16 streaming conversion (4 elems / thread) ----------------
__global__ void cvt_bf16_k(const float* __restrict__ x, unsigned short* __restrict__ y, int n4) {
    int i = blockIdx.x * blockDim.x + threadIdx.x;
    if (i >= n4) return;
    const float4 v = ((const float4*)x)[i];
    us4 o;
    o[0] = f2bf_u(v.x); o[1] = f2bf_u(v.y); o[2] = f2bf_u(v.z); o[3] = f2bf_u(v.w);
    ((us4*)y)[i] = o;
}

// ---------------- epilogue for one 16x16 tile ----------------
// ACT: 0=none 1=softplus 2=gelu(exact) 3=relu
template<int ACT, bool REV, bool ACC>
__device__ __forceinline__ void store_tile(const v8f& acc, int rbase, int col,
    const float* __restrict__ bias, const float* __restrict__ addsrc,
    const float* __restrict__ rowscale,
    float* __restrict__ outf, unsigned short* __restrict__ outb, int Nout, int L)
{
    if (col >= Nout) return;
    const float bval = bias ? bias[col] : 0.f;
#pragma unroll
    for (int r = 0; r < 8; ++r) {
        const int row = rbase + r;
        float v = acc[r] + bval;
        if (ACT == 1) v = (v > 20.f) ? v : log1pf(__expf(v));
        else if (ACT == 2) v = 0.5f * v * (1.f + erff(v * 0.70710678118654752f));
        else if (ACT == 3) v = fmaxf(v, 0.f);
        if (rowscale) v *= rowscale[row];
        if (addsrc)   v += addsrc[(size_t)row * Nout + col];
        size_t orow = row;
        if (REV) { int bb = row / L, ll = row - bb * L; orow = (size_t)bb * L + (L - 1 - ll); }
        if (ACC) outf[orow * Nout + col] += v;
        else {
            if (outf) outf[orow * Nout + col] = v;
            if (outb) outb[orow * Nout + col] = f2bf_u(v);
        }
    }
}

// ---------------- WMMA GEMM (bf16 in, f32 acc), MT=2 x NT tiles per wave ----------------
// outf[M,Nout] (f32, optional) / outb (bf16, optional) = act(A[M,K(lda)] * W[Nout,K]^T + bias)
// NGUARD=false requires all NT tiles fully in-range (N multiple of NT*16 or == NT*16).
template<int ACT, bool REV, bool ACC, int NT, bool NGUARD>
__global__ void wmma_gemm(const unsigned short* __restrict__ A, int lda,
                          const unsigned short* __restrict__ W,
                          const float* __restrict__ bias,
                          const float* __restrict__ addsrc,
                          const float* __restrict__ rowscale,
                          float* __restrict__ outf,
                          unsigned short* __restrict__ outb,
                          int M, int Nout, int K, int L)
{
    const int Ntiles = (Nout + 15) >> 4;
    const int Ngrp = (Ntiles + NT - 1) / NT;
    const int Mgrp = M >> 5;                 // MT=2
    const int wave = blockIdx.x * (blockDim.x >> 5) + (threadIdx.x >> 5);
    if (wave >= Mgrp * Ngrp) return;         // wave-uniform
    const int gm = wave / Ngrp, gn = wave % Ngrp;
    const int lane = threadIdx.x & 31;
    const int l15 = lane & 15;
    const int hi = lane >> 4;

    const unsigned short* A0 = A + (size_t)(gm * 32 + l15) * lda + (hi << 3);
    const unsigned short* A1 = A0 + (size_t)16 * lda;

    const unsigned short* Wp[NT];
    bool nv[NT];
#pragma unroll
    for (int t = 0; t < NT; ++t) {
        const int n = gn * (NT * 16) + t * 16 + l15;
        nv[t] = n < Nout;
        Wp[t] = W + (size_t)((NGUARD && !nv[t]) ? 0 : n) * K + (hi << 3);
    }

    const v8f zacc = {};
    v8f acc0[NT], acc1[NT];
#pragma unroll
    for (int t = 0; t < NT; ++t) { acc0[t] = zacc; acc1[t] = zacc; }

    const v16bf zf = {};
    for (int k0 = 0; k0 < K; k0 += 32) {
        const v16bf a0 = ld_frag_a(A0 + k0);
        const v16bf a1 = ld_frag_a(A1 + k0);
        v16bf bfrag[NT];
#pragma unroll
        for (int t = 0; t < NT; ++t) {
            v16bf bv = ld_frag_b(Wp[t] + k0);
            if (NGUARD && !nv[t]) bv = zf;
            bfrag[t] = bv;
        }
#pragma unroll
        for (int t = 0; t < NT; ++t) {
            acc0[t] = __builtin_amdgcn_wmma_f32_16x16x32_bf16(false, a0, false, bfrag[t], (short)0, acc0[t], false, false);
            acc1[t] = __builtin_amdgcn_wmma_f32_16x16x32_bf16(false, a1, false, bfrag[t], (short)0, acc1[t], false, false);
        }
    }

    // C/D layout: VGPR r -> row base + r (+8 for lanes 16-31), col = tile_n + (lane&15)
    const int rb0 = gm * 32 + (hi << 3);
#pragma unroll
    for (int t = 0; t < NT; ++t) {
        const int c = gn * (NT * 16) + t * 16 + l15;
        store_tile<ACT,REV,ACC>(acc0[t], rb0,      c, bias, addsrc, rowscale, outf, outb, Nout, L);
        store_tile<ACT,REV,ACC>(acc1[t], rb0 + 16, c, bias, addsrc, rowscale, outf, outb, Nout, L);
    }
}

// ---------------- LayerNorm over last dim D, one block per row; bf16 output ----------------
__global__ void layernorm_k(const float* __restrict__ x, const float* __restrict__ g,
                            const float* __restrict__ b, unsigned short* __restrict__ yb, int D)
{
    __shared__ float ws1[8], ws2[8];
    const int row = blockIdx.x;
    const float* xr = x + (size_t)row * D;
    float s = 0.f, s2 = 0.f;
    for (int j = threadIdx.x; j < D; j += blockDim.x) { float v = xr[j]; s += v; s2 += v * v; }
    for (int off = 16; off; off >>= 1) { s += __shfl_down(s, off, 32); s2 += __shfl_down(s2, off, 32); }
    const int wv = threadIdx.x >> 5;
    if ((threadIdx.x & 31) == 0) { ws1[wv] = s; ws2[wv] = s2; }
    __syncthreads();
    if (threadIdx.x == 0) {
        float t = 0.f, t2 = 0.f;
        for (int i = 0; i < (int)(blockDim.x >> 5); ++i) { t += ws1[i]; t2 += ws2[i]; }
        float mu = t / D;
        ws1[0] = mu;
        ws2[0] = rsqrtf(t2 / D - mu * mu + 1e-5f);
    }
    __syncthreads();
    const float mu = ws1[0], inv = ws2[0];
    unsigned short* yr = yb + (size_t)row * D;
    for (int j = threadIdx.x; j < D; j += blockDim.x)
        yr[j] = f2bf_u((xr[j] - mu) * inv * g[j] + b[j]);
}

// ---------------- depthwise causal conv (DC taps) + SiLU; dual f32+bf16 output ----------------
template<bool REVD>
__global__ void conv_silu_k(const float* __restrict__ xz,   // [B*L, 2*DI], xi = cols 0..DI-1
                            const float* __restrict__ cw,   // [DI, DC]
                            const float* __restrict__ cb,   // [DI]
                            float* __restrict__ xi,         // [B*L, DI] (dir orientation)
                            unsigned short* __restrict__ xib,
                            int B, int L, int DI, int DC)
{
    const int idx = blockIdx.x * blockDim.x + threadIdx.x;
    if (idx >= B * L * DI) return;
    const int c = idx % DI;
    const int t = idx / DI;
    const int l = t % L, bb = t / L;
    float a = cb[c];
    for (int j = 0; j < DC; ++j) {
        const int ls = l - (DC - 1) + j;
        if (ls < 0) continue;
        const int lsrc = REVD ? (L - 1 - ls) : ls;
        a += cw[c * DC + j] * xz[((size_t)bb * L + lsrc) * (size_t)(2 * DI) + c];
    }
    const float v = silu_f(a);
    xi[(size_t)t * DI + c] = v;
    xib[(size_t)t * DI + c] = f2bf_u(v);
}

// ---------------- selective scan: one thread per (b, d); 16 states in registers ----------------
template<bool REVD>
__global__ void scan_k(const float* __restrict__ xi,    // [B*L, DI] dir-oriented
                       const float* __restrict__ dt,    // [B*L, DI] dir-oriented (softplus'd)
                       const float* __restrict__ xdbl,  // [B*L, 64]; Bm cols 32..47, Cm 48..63
                       const float* __restrict__ alog,  // [DI,16]
                       const float* __restrict__ Dp,    // [DI]
                       const float* __restrict__ xz,    // [B*L, 2*DI]; z = cols DI.. (natural)
                       unsigned short* __restrict__ g,  // [B*L, DI] dir-oriented bf16: silu(z)*y
                       int B, int L, int DI)
{
    const int idx = blockIdx.x * blockDim.x + threadIdx.x;
    if (idx >= B * DI) return;
    const int d = idx % DI, bb = idx / DI;
    float Av[16], h[16];
#pragma unroll
    for (int s = 0; s < 16; ++s) { Av[s] = -__expf(alog[d * 16 + s]); h[s] = 0.f; }
    const float Dv = Dp[d];
    for (int l = 0; l < L; ++l) {
        const size_t r = (size_t)bb * L + l;
        const float dtv = dt[r * DI + d];
        const float xv  = xi[r * DI + d];
        const float* Bm = xdbl + r * 64 + 32;
        const float* Cm = xdbl + r * 64 + 48;
        float y = 0.f;
#pragma unroll
        for (int s = 0; s < 16; ++s) {
            h[s] = __expf(dtv * Av[s]) * h[s] + dtv * Bm[s] * xv;
            y += h[s] * Cm[s];
        }
        y += Dv * xv;
        const int lz = REVD ? (L - 1 - l) : l;
        const float zv = xz[((size_t)bb * L + lz) * (size_t)(2 * DI) + DI + d];
        g[r * DI + d] = f2bf_u(silu_f(zv) * y);
    }
}

// ---------------- MoE top-1 gate mass: scale[row] = max softmax(h@gw.T+gb) ----------------
__global__ void moe_gate_k(const float* __restrict__ h, const float* __restrict__ gw,
                           const float* __restrict__ gb, float* __restrict__ scale,
                           int M, int DM, int E)
{
    const int row = blockIdx.x * blockDim.x + threadIdx.x;
    if (row >= M) return;
    float logits[8];
    float mx = -1e30f;
    const float* hr = h + (size_t)row * DM;
    for (int e = 0; e < E; ++e) {
        float s = gb[e];
        const float* w = gw + (size_t)e * DM;
        for (int k = 0; k < DM; ++k) s += hr[k] * w[k];
        logits[e] = s;
        mx = fmaxf(mx, s);
    }
    float denom = 0.f;
    for (int e = 0; e < E; ++e) denom += __expf(logits[e] - mx);
    scale[row] = 1.f / denom;   // top-1 softmax mass
}

// ---------------- host orchestration ----------------
static dim3 gemm_grid(int M, int N, int NT) {
    int ngrp = (((N + 15) / 16) + NT - 1) / NT;
    int waves = (M / 32) * ngrp;
    return dim3((waves + 7) / 8);
}

extern "C" void kernel_launch(void* const* d_in, const int* in_sizes, int n_in,
                              void* d_out, int out_size, void* d_ws, size_t ws_size,
                              hipStream_t stream)
{
    (void)in_sizes; (void)n_in; (void)out_size; (void)ws_size;
    const float* x     = (const float*)d_in[0];
    const float* inp_w = (const float*)d_in[1];
    const float* inp_b = (const float*)d_in[2];
    const float* n1g   = (const float*)d_in[3];
    const float* n1b   = (const float*)d_in[4];
    const float* inw   = (const float*)d_in[5];
    const float* cw    = (const float*)d_in[6];
    const float* cb    = (const float*)d_in[7];
    const float* xpw   = (const float*)d_in[8];
    const float* dtw   = (const float*)d_in[9];
    const float* dtbp  = (const float*)d_in[10];
    const float* alog  = (const float*)d_in[11];
    const float* Dp    = (const float*)d_in[12];
    const float* ow    = (const float*)d_in[13];
    const float* n2g   = (const float*)d_in[14];
    const float* n2b   = (const float*)d_in[15];
    const float* fw1   = (const float*)d_in[16];
    const float* fb1   = (const float*)d_in[17];
    const float* fw2   = (const float*)d_in[18];
    const float* fb2   = (const float*)d_in[19];
    const float* gw    = (const float*)d_in[20];
    const float* gb    = (const float*)d_in[21];
    const float* ew1   = (const float*)d_in[22];
    const float* eb1   = (const float*)d_in[23];
    const float* ew2   = (const float*)d_in[24];
    const float* eb2   = (const float*)d_in[25];
    const float* ew3   = (const float*)d_in[26];
    const float* eb3   = (const float*)d_in[27];
    const float* fcw   = (const float*)d_in[28];
    const float* fcb   = (const float*)d_in[29];

    const int B = 4, L = 512, IN = 128, DM = 512, DI = 1024, DC = 4, DTR = 32, NL = 2, E = 8, NC = 10;
    const int M = B * L;            // 2048 token rows

    // ---- workspace carve-up (byte-based; everything stays 16B aligned) ----
    char* P = (char*)d_ws;
    size_t o = 0;
    auto alloc_f = [&](size_t n) { float* p = (float*)(P + o); o += n * 4; return p; };
    auto alloc_h = [&](size_t n) { unsigned short* p = (unsigned short*)(P + o); o += ((n + 7) & ~(size_t)7) * 2; return p; };

    float* hbuf  = alloc_f((size_t)M * DM);
    float* hbuf2 = alloc_f((size_t)M * DM);
    float* mam   = alloc_f((size_t)M * DM);
    float* xz    = alloc_f((size_t)M * 2 * DI);
    float* xi    = alloc_f((size_t)M * DI);
    float* dt_   = alloc_f((size_t)M * DI);
    float* xdbl  = alloc_f((size_t)M * 64);
    float* scale = alloc_f((size_t)M);

    unsigned short* x_bf    = alloc_h((size_t)M * IN);
    unsigned short* hn_bf   = alloc_h((size_t)M * DM);
    unsigned short* xi_bf   = alloc_h((size_t)M * DI);
    unsigned short* xdbl_bf = alloc_h((size_t)M * 64);
    unsigned short* g_bf    = alloc_h((size_t)M * DI);
    unsigned short* ff1_bf  = alloc_h((size_t)M * 2 * DI);
    unsigned short* h2_bf   = alloc_h((size_t)M * DM);
    unsigned short* h_bf    = alloc_h((size_t)M * DM);
    unsigned short* e1_bf   = alloc_h((size_t)M * 256);
    unsigned short* e2_bf   = alloc_h((size_t)M * 256);

    unsigned short* wb_inpw = alloc_h((size_t)DM * IN);
    unsigned short* wb_inw  = alloc_h((size_t)NL * 2 * DI * DM);
    unsigned short* wb_xpw  = alloc_h((size_t)NL * 64 * DI);
    unsigned short* wb_dtw  = alloc_h((size_t)NL * DI * DTR);
    unsigned short* wb_ow   = alloc_h((size_t)NL * DM * DI);
    unsigned short* wb_fw1  = alloc_h((size_t)NL * 4 * DM * DM);
    unsigned short* wb_fw2  = alloc_h((size_t)NL * DM * 4 * DM);
    unsigned short* wb_ew1  = alloc_h((size_t)256 * DM);
    unsigned short* wb_ew2  = alloc_h((size_t)256 * 256);
    unsigned short* wb_ew3  = alloc_h((size_t)DM * 256);
    unsigned short* wb_fcw  = alloc_h((size_t)NC * DM + 8);

    const dim3 blk(256);
    auto cvt = [&](const float* src, unsigned short* dst, size_t n) {
        int n4 = (int)(n / 4);
        cvt_bf16_k<<<(n4 + 255) / 256, blk, 0, stream>>>(src, dst, n4);
    };

    // ---- one-time bf16 conversion of input + all weights ----
    cvt(x,     x_bf,    (size_t)M * IN);
    cvt(inp_w, wb_inpw, (size_t)DM * IN);
    cvt(inw,   wb_inw,  (size_t)NL * 2 * DI * DM);
    cvt(xpw,   wb_xpw,  (size_t)NL * 64 * DI);
    cvt(dtw,   wb_dtw,  (size_t)NL * DI * DTR);
    cvt(ow,    wb_ow,   (size_t)NL * DM * DI);
    cvt(fw1,   wb_fw1,  (size_t)NL * 4 * DM * DM);
    cvt(fw2,   wb_fw2,  (size_t)NL * DM * 4 * DM);
    cvt(ew1,   wb_ew1,  (size_t)256 * DM);
    cvt(ew2,   wb_ew2,  (size_t)256 * 256);
    cvt(ew3,   wb_ew3,  (size_t)DM * 256);
    cvt(fcw,   wb_fcw,  (size_t)NC * DM);

    // h = x @ inp_w.T + inp_b
    wmma_gemm<0,false,false,4,false><<<gemm_grid(M, DM, 4), blk, 0, stream>>>(
        x_bf, IN, wb_inpw, inp_b, nullptr, nullptr, hbuf, nullptr, M, DM, IN, L);

    for (int i = 0; i < NL; ++i) {
        // hn = LN1(h)  (bf16 out; consumed only by GEMMs)
        layernorm_k<<<M, 256, 0, stream>>>(hbuf, n1g + i * DM, n1b + i * DM, hn_bf, DM);
        // xz = hn @ inw.T   (per-token; shared by both directions; f32 for conv/scan)
        wmma_gemm<0,false,false,4,false><<<gemm_grid(M, 2 * DI, 4), blk, 0, stream>>>(
            hn_bf, DM, wb_inw + (size_t)i * 2 * DI * DM, nullptr, nullptr, nullptr,
            xz, nullptr, M, 2 * DI, DM, L);

        for (int dir = 0; dir < 2; ++dir) {
            const int cgrid = (M * DI + 255) / 256;
            if (dir == 0)
                conv_silu_k<false><<<cgrid, blk, 0, stream>>>(xz, cw + (size_t)i * DI * DC, cb + i * DI, xi, xi_bf, B, L, DI, DC);
            else
                conv_silu_k<true ><<<cgrid, blk, 0, stream>>>(xz, cw + (size_t)i * DI * DC, cb + i * DI, xi, xi_bf, B, L, DI, DC);
            // xdbl = xi @ xpw.T   [M,64]  (f32 for scan B/C + bf16 for dt GEMM)
            wmma_gemm<0,false,false,4,false><<<gemm_grid(M, 64, 4), blk, 0, stream>>>(
                xi_bf, DI, wb_xpw + (size_t)i * 64 * DI, nullptr, nullptr, nullptr,
                xdbl, xdbl_bf, M, 64, DI, L);
            // dt = softplus(xdbl[:, :32] @ dtw.T + dtb)   (A row stride 64)
            wmma_gemm<1,false,false,4,false><<<gemm_grid(M, DI, 4), blk, 0, stream>>>(
                xdbl_bf, 64, wb_dtw + (size_t)i * DI * DTR, dtbp + i * DI, nullptr, nullptr,
                dt_, nullptr, M, DI, DTR, L);
            const int sgrid = (B * DI + 255) / 256;
            if (dir == 0) {
                scan_k<false><<<sgrid, blk, 0, stream>>>(
                    xi, dt_, xdbl, alog + (size_t)i * DI * 16, Dp + i * DI, xz, g_bf, B, L, DI);
                // mam = g @ ow.T
                wmma_gemm<0,false,false,4,false><<<gemm_grid(M, DM, 4), blk, 0, stream>>>(
                    g_bf, DI, wb_ow + (size_t)i * DM * DI, nullptr, nullptr, nullptr,
                    mam, nullptr, M, DM, DI, L);
            } else {
                scan_k<true ><<<sgrid, blk, 0, stream>>>(
                    xi, dt_, xdbl, alog + (size_t)i * DI * 16, Dp + i * DI, xz, g_bf, B, L, DI);
                // mam += reverse_L(g @ ow.T)
                wmma_gemm<0,true,true,4,false><<<gemm_grid(M, DM, 4), blk, 0, stream>>>(
                    g_bf, DI, wb_ow + (size_t)i * DM * DI, nullptr, nullptr, nullptr,
                    mam, nullptr, M, DM, DI, L);
            }
        }

        // m = LN2(f + bwd)
        layernorm_k<<<M, 256, 0, stream>>>(mam, n2g + i * DM, n2b + i * DM, hn_bf, DM);
        // ff1 = gelu(m @ fw1.T + fb1)   [M, 2048]  (bf16 only; consumed by next GEMM)
        wmma_gemm<2,false,false,4,false><<<gemm_grid(M, 4 * DM, 4), blk, 0, stream>>>(
            hn_bf, DM, wb_fw1 + (size_t)i * 4 * DM * DM, fb1 + (size_t)i * 4 * DM, nullptr, nullptr,
            nullptr, ff1_bf, M, 4 * DM, DM, L);
        // h2 = ff1 @ fw2.T + fb2 + residual(h)   (f32 for gate + bf16 for MoE GEMM)
        wmma_gemm<0,false,false,4,false><<<gemm_grid(M, DM, 4), blk, 0, stream>>>(
            ff1_bf, 4 * DM, wb_fw2 + (size_t)i * DM * 4 * DM, fb2 + i * DM, hbuf, nullptr,
            hbuf2, h2_bf, M, DM, 4 * DM, L);

        // MoE (experts share one MLP; top-1 gate mass collapses to a row scale)
        moe_gate_k<<<(M + 255) / 256, blk, 0, stream>>>(hbuf2, gw, gb, scale, M, DM, E);
        wmma_gemm<3,false,false,4,false><<<gemm_grid(M, 256, 4), blk, 0, stream>>>(
            h2_bf, DM, wb_ew1, eb1, nullptr, nullptr, nullptr, e1_bf, M, 256, DM, L);
        wmma_gemm<3,false,false,4,false><<<gemm_grid(M, 256, 4), blk, 0, stream>>>(
            e1_bf, 256, wb_ew2, eb2, nullptr, nullptr, nullptr, e2_bf, M, 256, 256, L);
        wmma_gemm<3,false,false,4,false><<<gemm_grid(M, DM, 4), blk, 0, stream>>>(
            e2_bf, 256, wb_ew3, eb3, nullptr, scale, hbuf, h_bf, M, DM, 256, L);
    }

    // out = h @ fcw.T + fcb   [M, 10] (N-guarded single-tile column)
    wmma_gemm<0,false,false,1,true><<<gemm_grid(M, NC, 1), blk, 0, stream>>>(
        h_bf, DM, wb_fcw, fcb, nullptr, nullptr, (float*)d_out, nullptr, M, NC, DM, L);
}
